// Attention_85770496901380
// MI455X (gfx1250) — matrix operand order repaired and checked by reading, not compiled, not run on hardware
//
#include <hip/hip_runtime.h>
#include <hip/hip_bf16.h>
#include <math.h>

// ---------------------------------------------------------------------------
// MI455X (gfx1250) attention layer: f16 WMMA GEMMs + fused RoPE + flash attn
// ---------------------------------------------------------------------------

typedef __attribute__((ext_vector_type(16))) _Float16 v16h;
typedef __attribute__((ext_vector_type(8)))  _Float16 v8h;
typedef __attribute__((ext_vector_type(8)))  float    v8f;

#define N_HEADS 16
#define DHEAD   64
#define SEQ     2048
#define EMB     1024
#define BATCH   2
#define MROWS   (BATCH * SEQ)   /* 4096 */

static __device__ __forceinline__ v8f wmma16x16x32(v16h a, v16h b, v8f c) {
    // (neg_a, A, neg_b, B, c_mod, C, reuse_a, reuse_b)
    return __builtin_amdgcn_wmma_f32_16x16x32_f16(false, a, false, b,
                                                  (short)0, c, false, false);
}

// Load a 16x32 f16 A fragment (row-major, leading dim ld, in f16 elements).
// ISA A-layout: lane holds row (lane&15); element i (0..7) = K hs*8+i,
// element i (8..15) = K 16+hs*8+(i-8), hs = lane>>4.
static __device__ __forceinline__ v16h load_a16(const _Float16* __restrict__ p, int ld) {
    const int lane = threadIdx.x & 31;
    const int r  = lane & 15;
    const int g  = (lane >> 4) << 3;            // 0 or 8
    const _Float16* row = p + (size_t)r * ld;
    v8h lo = *(const v8h*)(row + g);
    v8h hi = *(const v8h*)(row + 16 + g);
    v16h out;
#pragma unroll
    for (int i = 0; i < 8; ++i) { out[i] = lo[i]; out[i + 8] = hi[i]; }
    return out;
}

// Load a 32x16 f16 B fragment from B^T (N x K row-major, leading dim ld).
// ISA B-layout: lane holds column (lane&15); element i (0..7) = K hs*8+i,
// element i (8..15) = K 16+hs*8+(i-8), hs = lane>>4.
static __device__ __forceinline__ v16h load_b16(const _Float16* __restrict__ pT, int ld) {
    const int lane = threadIdx.x & 31;
    const int c  = lane & 15;
    const int kh = (lane >> 4) << 3;            // 0 or 8
    const _Float16* row = pT + (size_t)c * ld + kh;
    v8h lo = *(const v8h*)(row);
    v8h hi = *(const v8h*)(row + 16);
    v16h out;
#pragma unroll
    for (int i = 0; i < 8; ++i) { out[i] = lo[i]; out[i + 8] = hi[i]; }
    return out;
}

// ---------------------------------------------------------------------------
// Conversions
// ---------------------------------------------------------------------------
__global__ void k_cvt_f32_to_f16(const float* __restrict__ src,
                                 _Float16* __restrict__ dst, int n) {
    int i = blockIdx.x * blockDim.x + threadIdx.x;
    if (i < n) dst[i] = (_Float16)src[i];
}

// src: R x C row-major f32 -> dst: C x R row-major f16 (dst[c][r] = src[r][c])
__global__ void k_transpose_to_f16(const float* __restrict__ src,
                                   _Float16* __restrict__ dst, int R, int C) {
    int i = blockIdx.x * blockDim.x + threadIdx.x;
    if (i < R * C) {
        int c = i / R, r = i - c * R;
        dst[i] = (_Float16)src[(size_t)r * C + c];
    }
}

// ---------------------------------------------------------------------------
// WMMA GEMM:  Y(MxN) = A(MxK) * B(KxN), B given transposed (BT: N x K).
// mode 0: RoPE epilogue, write f16 [B, H, S, D]      (Q and K)
// mode 1: plain epilogue, write f16 [B, H, D, S]     (V transposed)
// mode 2: plain epilogue, write f32 row-major M x N  (final projection)
// 256 threads = 8 waves; wave tile 32x32 (2x2 WMMA); WG tile 128x64.
// ---------------------------------------------------------------------------
__global__ void k_gemm_wmma(const _Float16* __restrict__ A,
                            const _Float16* __restrict__ BT,
                            _Float16* __restrict__ outH,
                            float* __restrict__ outF,
                            int M, int K, int N, int mode) {
    const int lane = threadIdx.x & 31;
    const int wave = threadIdx.x >> 5;
    const int wm = wave >> 1, wn = wave & 1;
    const int mbase = blockIdx.y * 128 + wm * 32;
    const int nbase = blockIdx.x * 64 + wn * 32;

    const v8f z8 = {0.f, 0.f, 0.f, 0.f, 0.f, 0.f, 0.f, 0.f};
    v8f acc[2][2] = {{z8, z8}, {z8, z8}};

    for (int kk = 0; kk < K; kk += 32) {
        v16h a0 = load_a16(A + (size_t)mbase * K + kk, K);
        v16h a1 = load_a16(A + (size_t)(mbase + 16) * K + kk, K);
        v16h b0 = load_b16(BT + (size_t)nbase * K + kk, K);
        v16h b1 = load_b16(BT + (size_t)(nbase + 16) * K + kk, K);
        acc[0][0] = wmma16x16x32(a0, b0, acc[0][0]);
        acc[0][1] = wmma16x16x32(a0, b1, acc[0][1]);
        acc[1][0] = wmma16x16x32(a1, b0, acc[1][0]);
        acc[1][1] = wmma16x16x32(a1, b1, acc[1][1]);
    }

    // C layout: VGPR j -> row j + 8*(lane>>4), col lane&15
    const int hs = lane >> 4;
    const int cn = lane & 15;
#pragma unroll
    for (int mi = 0; mi < 2; ++mi) {
#pragma unroll
        for (int ni = 0; ni < 2; ++ni) {
#pragma unroll
            for (int j = 0; j < 8; ++j) {
                int m = mbase + mi * 16 + j + 8 * hs;
                int n = nbase + ni * 16 + cn;
                float v = acc[mi][ni][j];
                if (mode == 2) {
                    outF[(size_t)m * N + n] = v;
                } else {
                    int b = m >> 11, s = m & (SEQ - 1);    // SEQ = 2048
                    int h = n >> 6,  d = n & (DHEAD - 1);  // DHEAD = 64
                    if (mode == 0) {
                        // RoPE: pair partner lives in adjacent lane (n parity == lane parity)
                        float partner = __shfl_xor(v, 1, 32);
                        int   fi = d >> 1;
                        // inv_freq = 10000^(-2*fi/64) = exp(-fi * ln(1e4)/32)
                        float inv_freq = __expf(-0.28782313662425572f * (float)fi);
                        float ang = (float)s * inv_freq;
                        float cs = __cosf(ang), sn = __sinf(ang);
                        float r = (d & 1) ? (partner * sn + v * cs)
                                          : (v * cs - partner * sn);
                        outH[((size_t)(b * N_HEADS + h) * SEQ + s) * DHEAD + d] = (_Float16)r;
                    } else { // mode 1: V stored transposed [B,H,D,S]
                        outH[((size_t)(b * N_HEADS + h) * DHEAD + d) * SEQ + s] = (_Float16)v;
                    }
                }
            }
        }
    }
}

// ---------------------------------------------------------------------------
// Flash attention: one wave per 16-query tile, 32 keys per iteration.
// Qh/Kh: [B,H,S,D] f16 (RoPE applied). VhT: [B,H,D,S] f16.
// Oh out: [B*S, EMB] f16 (heads re-interleaved for the final projection).
// ---------------------------------------------------------------------------
__global__ void k_flash_attn(const _Float16* __restrict__ Qh,
                             const _Float16* __restrict__ Kh,
                             const _Float16* __restrict__ VhT,
                             _Float16* __restrict__ Oh) {
    __shared__ __align__(16) _Float16 pbuf[8 * 16 * 32]; // 8 waves x (16x32) f16
    const int lane = threadIdx.x & 31;
    const int wave = threadIdx.x >> 5;
    const int task = blockIdx.x * 8 + wave;               // 0 .. B*H*(S/16)-1
    if (task >= BATCH * N_HEADS * (SEQ / 16)) return;
    const int qt = task & 127;                            // S/16 = 128
    const int bh = task >> 7;
    const int b  = bh >> 4, h = bh & 15;
    const int qbase = qt * 16;

    const _Float16* Qp = Qh + (size_t)bh * SEQ * DHEAD;
    const _Float16* Kp = Kh + (size_t)bh * SEQ * DHEAD;
    const _Float16* Vp = VhT + (size_t)bh * DHEAD * SEQ;
    _Float16* pw = pbuf + wave * (16 * 32);

    const v16h qa0 = load_a16(Qp + (size_t)qbase * DHEAD, DHEAD);
    const v16h qa1 = load_a16(Qp + (size_t)qbase * DHEAD + 32, DHEAD);

    const v8f z8 = {0.f, 0.f, 0.f, 0.f, 0.f, 0.f, 0.f, 0.f};
    v8f o[4] = {z8, z8, z8, z8};
    float mrun[8], lrun[8];
#pragma unroll
    for (int j = 0; j < 8; ++j) { mrun[j] = -__builtin_inff(); lrun[j] = 0.f; }

    const int hs = lane >> 4, cn = lane & 15;
    const float scale = 0.125f;                           // 1/sqrt(64)
    const int nkb = (qbase + 16 + 31) >> 5;               // key blocks of 32

    for (int kb = 0; kb < nkb; ++kb) {
        const int kbase = kb * 32;
        v8f sc[2];
#pragma unroll
        for (int ct = 0; ct < 2; ++ct) {
            v16h kf0 = load_b16(Kp + (size_t)(kbase + ct * 16) * DHEAD + 0, DHEAD);
            v16h kf1 = load_b16(Kp + (size_t)(kbase + ct * 16) * DHEAD + 32, DHEAD);
            v8f s = z8;
            s = wmma16x16x32(qa0, kf0, s);
            s = wmma16x16x32(qa1, kf1, s);
            sc[ct] = s;
        }
        // scale + causal mask
#pragma unroll
        for (int ct = 0; ct < 2; ++ct)
#pragma unroll
            for (int j = 0; j < 8; ++j) {
                int q   = qbase + j + 8 * hs;
                int key = kbase + ct * 16 + cn;
                float v = sc[ct][j] * scale;
                sc[ct][j] = (key > q) ? -__builtin_inff() : v;
            }
        // online softmax (row stats shared across each 16-lane half)
#pragma unroll
        for (int j = 0; j < 8; ++j) {
            float tmax = fmaxf(sc[0][j], sc[1][j]);
#pragma unroll
            for (int off = 1; off < 16; off <<= 1)
                tmax = fmaxf(tmax, __shfl_xor(tmax, off, 32));
            float mnew  = fmaxf(mrun[j], tmax);
            float alpha = __expf(mrun[j] - mnew);          // exp(-inf) = 0 first pass
            float p0 = __expf(sc[0][j] - mnew);
            float p1 = __expf(sc[1][j] - mnew);
            float rs = p0 + p1;
#pragma unroll
            for (int off = 1; off < 16; off <<= 1)
                rs += __shfl_xor(rs, off, 32);
            lrun[j] = lrun[j] * alpha + rs;
            mrun[j] = mnew;
            int row = j + 8 * hs;
            pw[row * 32 + cn]      = (_Float16)p0;
            pw[row * 32 + 16 + cn] = (_Float16)p1;
#pragma unroll
            for (int nt = 0; nt < 4; ++nt) o[nt][j] *= alpha;
        }
        // reload P with A-fragment layout from LDS (ld = 32)
        v16h pa = load_a16(pw, 32);
        // P(16x32) x V(32x64): V^T rows are d, contiguous in keys
#pragma unroll
        for (int nt = 0; nt < 4; ++nt) {
            v16h vf = load_b16(Vp + (size_t)(nt * 16) * SEQ + kbase, SEQ);
            o[nt] = wmma16x16x32(pa, vf, o[nt]);
        }
    }

    // normalize + scatter to [B*S, EMB] for the output projection
#pragma unroll
    for (int j = 0; j < 8; ++j) {
        float inv = 1.0f / lrun[j];
        int row = qbase + j + 8 * hs;
        size_t ob = ((size_t)(b * SEQ + row)) * EMB + h * DHEAD;
#pragma unroll
        for (int nt = 0; nt < 4; ++nt)
            Oh[ob + nt * 16 + cn] = (_Float16)(o[nt][j] * inv);
    }
}

// ---------------------------------------------------------------------------
// Host launcher
// ---------------------------------------------------------------------------
extern "C" void kernel_launch(void* const* d_in, const int* in_sizes, int n_in,
                              void* d_out, int out_size, void* d_ws, size_t ws_size,
                              hipStream_t stream) {
    const float* x  = (const float*)d_in[0];
    const float* Wq = (const float*)d_in[1];
    const float* Wk = (const float*)d_in[2];
    const float* Wv = (const float*)d_in[3];
    const float* Wo = (const float*)d_in[4];
    float* out = (float*)d_out;

    char* ws = (char*)d_ws;
    size_t off = 0;
    _Float16* xh  = (_Float16*)(ws + off); off += (size_t)MROWS * EMB * 2;   // 8 MiB
    _Float16* WqT = (_Float16*)(ws + off); off += (size_t)EMB * EMB * 2;     // 2 MiB
    _Float16* WkT = (_Float16*)(ws + off); off += (size_t)EMB * EMB * 2;
    _Float16* WvT = (_Float16*)(ws + off); off += (size_t)EMB * EMB * 2;
    _Float16* WoT = (_Float16*)(ws + off); off += (size_t)EMB * EMB * 2;
    _Float16* Qh  = (_Float16*)(ws + off); off += (size_t)MROWS * EMB * 2;   // [B,H,S,D]
    _Float16* Kh  = (_Float16*)(ws + off); off += (size_t)MROWS * EMB * 2;   // [B,H,S,D]
    _Float16* VhT = (_Float16*)(ws + off); off += (size_t)MROWS * EMB * 2;   // [B,H,D,S]
    _Float16* Oh  = (_Float16*)(ws + off); off += (size_t)MROWS * EMB * 2;   // [B*S, EMB]

    const int nX = MROWS * EMB;
    k_cvt_f32_to_f16<<<(nX + 255) / 256, 256, 0, stream>>>(x, xh, nX);

    const int nW = EMB * EMB;
    k_transpose_to_f16<<<(nW + 255) / 256, 256, 0, stream>>>(Wq, WqT, EMB, EMB);
    k_transpose_to_f16<<<(nW + 255) / 256, 256, 0, stream>>>(Wk, WkT, EMB, EMB);
    k_transpose_to_f16<<<(nW + 255) / 256, 256, 0, stream>>>(Wv, WvT, EMB, EMB);
    k_transpose_to_f16<<<(nW + 255) / 256, 256, 0, stream>>>(Wo, WoT, EMB, EMB);

    dim3 g(EMB / 64, MROWS / 128);
    // Q, K projections with fused RoPE; V projection stored transposed
    k_gemm_wmma<<<g, 256, 0, stream>>>(xh, WqT, Qh, nullptr, MROWS, EMB, EMB, 0);
    k_gemm_wmma<<<g, 256, 0, stream>>>(xh, WkT, Kh, nullptr, MROWS, EMB, EMB, 0);
    k_gemm_wmma<<<g, 256, 0, stream>>>(xh, WvT, VhT, nullptr, MROWS, EMB, EMB, 1);

    // flash attention: B*H*(S/16) = 4096 wave tasks, 8 waves per block
    k_flash_attn<<<(BATCH * N_HEADS * (SEQ / 16)) / 8, 256, 0, stream>>>(Qh, Kh, VhT, Oh);

    // output projection -> fp32
    k_gemm_wmma<<<g, 256, 0, stream>>>(Oh, WoT, nullptr, out, MROWS, EMB, EMB, 2);
}
